// MultiheadCrossAttention_25709674234096
// MI455X (gfx1250) — matrix operand order repaired and checked by reading, not compiled, not run on hardware
//
#include <hip/hip_runtime.h>
#include <hip/hip_bf16.h>

typedef __bf16 bf16;
typedef __attribute__((ext_vector_type(16))) __bf16 v16bf;
typedef __attribute__((ext_vector_type(8)))  __bf16 v8bf;
typedef __attribute__((ext_vector_type(8)))  float  v8f;
typedef __attribute__((ext_vector_type(4)))  int    v4i;

// ---- gfx1250 async global->LDS copy (ASYNCcnt) with sync fallback ----------
#if defined(__has_builtin)
#if __has_builtin(__builtin_amdgcn_global_load_async_to_lds_b128) && \
    __has_builtin(__builtin_amdgcn_s_wait_asynccnt)
#define USE_ASYNC_LDS 1
#endif
#endif

// copy 16 bytes (8 bf16) from global to LDS
__device__ __forceinline__ void copy_b128(const bf16* g, bf16* l) {
#ifdef USE_ASYNC_LDS
    __builtin_amdgcn_global_load_async_to_lds_b128(
        (__attribute__((address_space(1))) v4i*)g,
        (__attribute__((address_space(3))) v4i*)l, 0, 0);
#else
    *(v8bf*)l = *(const v8bf*)g;
#endif
}
__device__ __forceinline__ void wait_async_copies() {
#ifdef USE_ASYNC_LDS
    __builtin_amdgcn_s_wait_asynccnt(0);
#endif
}

// ---- fragment loaders -------------------------------------------------------
// A fragment (16x32 bf16): lane holds K contiguous kb..kb+15, kb=(lane>>4)*16.
__device__ __forceinline__ v16bf load_fragA(const bf16* p) {
    union { v16bf v; v8bf h[2]; } u;
    u.h[0] = *(const v8bf*)p;
    u.h[1] = *(const v8bf*)(p + 8);
    return u.v;
}
// B fragment (32x16 bf16): lane holds column (lane&15), K contiguous
// (lane>>4)*16 .. +15.  32 contiguous bytes at p.
__device__ __forceinline__ v16bf load_fragB(const bf16* p) {
    union { v16bf v; v8bf h[2]; } u;
    u.h[0] = *(const v8bf*)p;
    u.h[1] = *(const v8bf*)(p + 8);
    return u.v;
}

__device__ __forceinline__ v8bf load8_cvt(const float* p) {
    v8bf t;
    #pragma unroll
    for (int i = 0; i < 8; i++) t[i] = (bf16)p[i];
    return t;
}

// ---- one-shot: W[K][N] f32 -> Wt[N][K] bf16 --------------------------------
__global__ __launch_bounds__(256)
void transpose_cvt_kernel(const float* __restrict__ W, bf16* __restrict__ Wt,
                          int K, int N)
{
    __shared__ bf16 tile[64][65];
    const int k0 = blockIdx.x * 64;
    const int n0 = blockIdx.y * 64;
    const int tid = threadIdx.x;
    const int r  = tid >> 2;          // 0..63
    const int c0 = (tid & 3) * 16;    // 0,16,32,48
    {
        const float* src = W + (size_t)(k0 + r) * N + n0 + c0;
        #pragma unroll
        for (int i = 0; i < 16; i++) tile[r][c0 + i] = (bf16)src[i];
    }
    __syncthreads();
    {
        v8bf o0, o1;
        #pragma unroll
        for (int i = 0; i < 8; i++) o0[i] = tile[c0 + i][r];
        #pragma unroll
        for (int i = 0; i < 8; i++) o1[i] = tile[c0 + 8 + i][r];
        bf16* dst = Wt + (size_t)(n0 + r) * K + k0 + c0;
        *(v8bf*)dst       = o0;
        *(v8bf*)(dst + 8) = o1;
    }
}

// ---- GEMM: C[M,N] = A[M,K] @ W + bias;  W given pre-transposed Wt[N][K] ----
// Block: 256 threads (8 waves), tile BM=128, BN=128, BK=32, double-buffered.
// Waves 4(M, 32 rows each) x 2(N, 64 cols each); 8 WMMAs / wave / k-step.
// MODE 0: f32 out.  MODE 1: bf16 out (scaled).  MODE 2: kv split
//   (M = 2*1024 rows = (b,s); wn==0 cols -> K[b,h,s,c]; wn==1 -> V^T[b,h,c,s])
template <typename AT, int MODE>
__global__ __launch_bounds__(256)
void gemm_kernel(const AT* __restrict__ A, const bf16* __restrict__ Wt,
                 const float* __restrict__ bias, float outScale,
                 float* __restrict__ Cf, bf16* __restrict__ Cb,
                 bf16* __restrict__ Kb, bf16* __restrict__ VTb,
                 int M, int N, int K)
{
    __shared__ bf16 As[2][128][32];      // [m][k]
    __shared__ bf16 Ws[2][128][40];      // [n][k], padded stride 40

    const int tid  = threadIdx.x;
    const int lane = tid & 31;
    const int wave = tid >> 5;
    const int wm   = wave & 3;           // 32-row sub-block
    const int wn   = wave >> 2;          // 64-col sub-block
    const int m0   = blockIdx.x * 128;
    const int n0   = blockIdx.y * 128;

    const int lrow = lane & 15;
    const int kbA  = (lane >> 4) * 16;
    const int kbB  = (lane >> 4) * 16;

    // cooperative-load indices: 256 threads, 16 halves each per tile
    const int ar = tid >> 1;             // 0..127
    const int ac = (tid & 1) * 16;       // 0,16

    v8f acc[2][4] = {};

    auto load_tiles = [&](int k0, int buf) {
        // A tile 128x32
        const AT* asrc = A + (size_t)(m0 + ar) * K + k0 + ac;
        if constexpr (sizeof(AT) == 2) {
            copy_b128((const bf16*)asrc,     &As[buf][ar][ac]);
            copy_b128((const bf16*)asrc + 8, &As[buf][ar][ac + 8]);
        } else {
            *(v8bf*)&As[buf][ar][ac]     = load8_cvt((const float*)asrc);
            *(v8bf*)&As[buf][ar][ac + 8] = load8_cvt((const float*)asrc + 8);
            __builtin_prefetch(asrc + 32, 0, 0);
        }
        // B tile 128x32 from pre-transposed weights: pure bf16 copy
        const bf16* bsrc = Wt + (size_t)(n0 + ar) * K + k0 + ac;
        copy_b128(bsrc,     &Ws[buf][ar][ac]);
        copy_b128(bsrc + 8, &Ws[buf][ar][ac + 8]);
    };

    load_tiles(0, 0);
    int it = 0;
    for (int k0 = 0; k0 < K; k0 += 32, it++) {
        const int buf = it & 1;
        wait_async_copies();
        __syncthreads();
        if (k0 + 32 < K) load_tiles(k0 + 32, buf ^ 1);

        v16bf bfrag[4];
        #pragma unroll
        for (int j = 0; j < 4; j++)
            bfrag[j] = load_fragB(&Ws[buf][wn * 64 + j * 16 + lrow][kbB]);
        #pragma unroll
        for (int sub = 0; sub < 2; sub++) {
            const v16bf af =
                load_fragA(&As[buf][wm * 32 + sub * 16 + lrow][kbA]);
            #pragma unroll
            for (int j = 0; j < 4; j++)
                acc[sub][j] = __builtin_amdgcn_wmma_f32_16x16x32_bf16(
                    false, af, false, bfrag[j], (short)0, acc[sub][j], false, false);
        }
    }

    // store: row = m0 + wm*32 + sub*16 + v + 8*hi, col = n0 + wn*64 + j*16 + lrow
    const int hi = lane >> 4;
    #pragma unroll
    for (int sub = 0; sub < 2; sub++) {
        #pragma unroll
        for (int j = 0; j < 4; j++) {
            const int col = n0 + wn * 64 + j * 16 + lrow;
            const float bv = bias[col];
            #pragma unroll
            for (int v = 0; v < 8; v++) {
                const int row = m0 + wm * 32 + sub * 16 + v + 8 * hi;
                const float val = (acc[sub][j][v] + bv) * outScale;
                if (MODE == 0) {
                    Cf[(size_t)row * N + col] = val;
                } else if (MODE == 1) {
                    Cb[(size_t)row * N + col] = (bf16)val;
                } else {
                    // kv split: rem<64 iff wn==0 (wave-uniform branch)
                    const int h   = col >> 7;
                    const int rem = col & 127;
                    const int bb  = row >> 10;      // M = 2*1024
                    const int s   = row & 1023;
                    if (wn == 0)
                        Kb[(((size_t)bb * 16 + h) * 1024 + s) * 64 + rem] = (bf16)val;
                    else
                        VTb[(((size_t)bb * 16 + h) * 64 + (rem - 64)) * 1024 + s] = (bf16)val;
                }
            }
        }
    }
}

// ---- Flash attention: per (b, h, 64 query rows); 4 waves x 16 rows --------
// q:  [b, 4096, 1024] bf16 (scale^2 pre-folded)
// Kb: [b, h, 1024, 64] bf16      VTb: [b, h, 64, 1024] bf16
// out:[b, 4096, 1024] bf16
__global__ __launch_bounds__(128)
void attn_kernel(const bf16* __restrict__ q, const bf16* __restrict__ Kb,
                 const bf16* __restrict__ VTb, bf16* __restrict__ out)
{
    constexpr int WID = 1024, NCTX = 4096, NDATA = 1024, CH = 64;
    constexpr int NSB = NDATA / 64;       // 16 s-blocks
    const int tb = blockIdx.x, h = blockIdx.y, b = blockIdx.z;
    const int tid  = threadIdx.x;
    const int lane = tid & 31;
    const int wave = tid >> 5;            // 0..3
    const int lrow = lane & 15;
    const int hi   = lane >> 4;
    const int kbA  = hi * 16;
    const int kbB  = hi * 16;

    __shared__ bf16 Ks[2][64][64];        // [s][c]  (B frag source for Q@K^T)
    __shared__ bf16 Vt[2][64][64];        // [c][s]  (B frag source for P@V)
    __shared__ bf16 Ps[4][16][64];        // per-wave P scratch [row][s]

    const bf16* kbase  = Kb  + (((size_t)b * 16 + h) * NDATA) * CH;
    const bf16* vtbase = VTb + (((size_t)b * 16 + h) * CH) * NDATA;

    const int cr = tid >> 3;              // 0..15 (+16*pass)
    const int c8 = (tid & 7) * 8;         // 0..56

    auto load_tile = [&](int sb2, int buf2) {
        const int s0 = sb2 * 64;
        #pragma unroll
        for (int pass = 0; pass < 4; pass++) {
            const int r = cr + pass * 16;
            copy_b128(kbase + (size_t)(s0 + r) * CH + c8, &Ks[buf2][r][c8]);
            copy_b128(vtbase + (size_t)r * NDATA + s0 + c8, &Vt[buf2][r][c8]);
        }
    };

    // Q fragments: 16 rows x 64ch -> 2 A-frags, held in registers
    const int trow = tb * 64 + wave * 16 + lrow;
    const bf16* qrow = q + ((size_t)b * NCTX + trow) * WID + h * CH;
    v16bf qa[2];
    qa[0] = load_fragA(qrow + kbA);
    qa[1] = load_fragA(qrow + 32 + kbA);

    float m_run[8], l_run[8];
    v8f O[4] = {};
    #pragma unroll
    for (int v = 0; v < 8; v++) { m_run[v] = -1e30f; l_run[v] = 0.f; }

    load_tile(0, 0);
    for (int sb = 0; sb < NSB; sb++) {
        const int buf = sb & 1;
        wait_async_copies();
        __syncthreads();
        if (sb + 1 < NSB) load_tile(sb + 1, buf ^ 1);

        // S = Q @ K^T  (16 x 64 per wave)
        v8f S[4] = {};
        #pragma unroll
        for (int j = 0; j < 4; j++) {
            const v16bf bk0 = load_fragB(&Ks[buf][j * 16 + lrow][kbB]);
            const v16bf bk1 = load_fragB(&Ks[buf][j * 16 + lrow][32 + kbB]);
            S[j] = __builtin_amdgcn_wmma_f32_16x16x32_bf16(
                false, qa[0], false, bk0, (short)0, S[j], false, false);
            S[j] = __builtin_amdgcn_wmma_f32_16x16x32_bf16(
                false, qa[1], false, bk1, (short)0, S[j], false, false);
        }

        // online softmax
        float alpha[8];
        #pragma unroll
        for (int v = 0; v < 8; v++) {
            float mv = fmaxf(fmaxf(S[0][v], S[1][v]), fmaxf(S[2][v], S[3][v]));
            #pragma unroll
            for (int off = 1; off < 16; off <<= 1)
                mv = fmaxf(mv, __shfl_xor(mv, off, 32));
            const float mn = fmaxf(m_run[v], mv);
            alpha[v] = __expf(m_run[v] - mn);
            m_run[v] = mn;
        }
        #pragma unroll
        for (int v = 0; v < 8; v++) {
            float rs = 0.f;
            #pragma unroll
            for (int j = 0; j < 4; j++) {
                const float p = __expf(S[j][v] - m_run[v]);
                S[j][v] = p;
                rs += p;
            }
            #pragma unroll
            for (int off = 1; off < 16; off <<= 1) rs += __shfl_xor(rs, off, 32);
            l_run[v] = l_run[v] * alpha[v] + rs;
            #pragma unroll
            for (int j = 0; j < 4; j++) O[j][v] *= alpha[v];
        }

        // P (C layout) -> per-wave LDS scratch (A layout source)
        #pragma unroll
        for (int j = 0; j < 4; j++)
            #pragma unroll
            for (int v = 0; v < 8; v++)
                Ps[wave][v + 8 * hi][j * 16 + lrow] = (bf16)S[j][v];
        asm volatile("s_wait_dscnt 0" ::: "memory");

        // O += P @ V
        #pragma unroll
        for (int kc = 0; kc < 2; kc++) {
            const v16bf pa = load_fragA(&Ps[wave][lrow][kc * 32 + kbA]);
            #pragma unroll
            for (int j = 0; j < 4; j++) {
                const v16bf vb = load_fragB(&Vt[buf][j * 16 + lrow][kc * 32 + kbB]);
                O[j] = __builtin_amdgcn_wmma_f32_16x16x32_bf16(
                    false, pa, false, vb, (short)0, O[j], false, false);
            }
        }
    }

    // normalize + store [b, t, h*64 + c] bf16
    #pragma unroll
    for (int j = 0; j < 4; j++)
        #pragma unroll
        for (int v = 0; v < 8; v++) {
            const int row = tb * 64 + wave * 16 + v + 8 * hi;
            out[((size_t)b * NCTX + row) * WID + h * CH + j * 16 + lrow] =
                (bf16)(O[j][v] / l_run[v]);
        }
}

// ---------------------------------------------------------------------------
extern "C" void kernel_launch(void* const* d_in, const int* in_sizes, int n_in,
                              void* d_out, int out_size, void* d_ws, size_t ws_size,
                              hipStream_t stream)
{
    (void)in_sizes; (void)n_in; (void)out_size; (void)ws_size;
    const float* x    = (const float*)d_in[0];  // [2,4096,1024]
    const float* data = (const float*)d_in[1];  // [2,1024,1024]
    const float* Wq   = (const float*)d_in[2];  // [1024,1024]
    const float* bq   = (const float*)d_in[3];
    const float* Wkv  = (const float*)d_in[4];  // [1024,2048]
    const float* bkv  = (const float*)d_in[5];
    const float* Wp   = (const float*)d_in[6];  // [1024,1024]
    const float* bp   = (const float*)d_in[7];
    float* outp = (float*)d_out;                // [2,4096,1024]

    const int BS = 2, NCTX = 4096, NDATA = 1024, W = 1024, CH = 64, H = 16;
    const int Mq  = BS * NCTX;    // 8192
    const int Mkv = BS * NDATA;   // 2048

    bf16* qb    = (bf16*)d_ws;                        // 8192*1024
    bf16* attnb = qb    + (size_t)Mq * W;             // 8192*1024
    bf16* Kbuf  = attnb + (size_t)Mq * W;             // 2*16*1024*64
    bf16* VTbuf = Kbuf  + (size_t)BS * H * NDATA * CH;// 2*16*64*1024
    bf16* Wqt   = VTbuf + (size_t)BS * H * CH * NDATA;// [1024][1024]
    bf16* Wkvt  = Wqt   + (size_t)W * W;              // [2048][1024]
    bf16* Wpt   = Wkvt  + (size_t)(2 * W) * W;        // [1024][1024]

    const float scale2 = 0.125f;  // (ch^-0.25)^2 = 1/sqrt(64), folded into q

    // 0) weight transpose+convert (one-shot, ~8 MB total)
    transpose_cvt_kernel<<<dim3(W / 64, W / 64), 256, 0, stream>>>(Wq, Wqt, W, W);
    transpose_cvt_kernel<<<dim3(W / 64, (2 * W) / 64), 256, 0, stream>>>(Wkv, Wkvt, W, 2 * W);
    transpose_cvt_kernel<<<dim3(W / 64, W / 64), 256, 0, stream>>>(Wp, Wpt, W, W);

    // 1) q = (x @ Wq + bq) * scale^2 -> bf16
    gemm_kernel<float, 1><<<dim3(Mq / 128, W / 128), 256, 0, stream>>>(
        x, Wqt, bq, scale2, nullptr, qb, nullptr, nullptr, Mq, W, W);

    // 2) kv = data @ Wkv + bkv -> K[b,h,s,c], V^T[b,h,c,s] bf16
    gemm_kernel<float, 2><<<dim3(Mkv / 128, (2 * W) / 128), 256, 0, stream>>>(
        data, Wkvt, bkv, 1.0f, nullptr, nullptr, Kbuf, VTbuf, Mkv, 2 * W, W);

    // 3) attention -> bf16 [b, t, width]
    attn_kernel<<<dim3(NCTX / 64, H, BS), 128, 0, stream>>>(qb, Kbuf, VTbuf, attnb);

    // 4) out = attn @ Wp + bp -> f32
    gemm_kernel<bf16, 0><<<dim3(Mq / 128, W / 128), 256, 0, stream>>>(
        attnb, Wpt, bp, 1.0f, outp, nullptr, nullptr, nullptr, Mq, W, W);
}
